// net_stratified_norm_85710367359314
// MI455X (gfx1250) — compile-verified, matrix operand order checked by reading, not run on hardware
//
#include <hip/hip_runtime.h>

// ---------------------------------------------------------------------------
// MI455X (gfx1250) fused stratified-norm MLP.
// Bandwidth-bound design (~0.91 GB min traffic => ~40us @ 23.3 TB/s); GEMMs on
// v_wmma_f32_16x16x32_f16 (wave32 WMMA), activations stored f16, group stats
// fused into the GEMM epilogue via LDS float atomics + global float atomics.
// ---------------------------------------------------------------------------

typedef __attribute__((ext_vector_type(16))) _Float16 v16h;
typedef __attribute__((ext_vector_type(8)))  float    v8f;

#define GROUPS 32
#define HID    64
#define DIN    248
#define EPSI   1e-8f

__device__ __forceinline__ float lrelu(float v) { return v >= 0.f ? v : 0.01f * v; }

// ---------------------------------------------------------------------------
__global__ void k_zero(float* p, int n) {
    for (int i = blockIdx.x * blockDim.x + threadIdx.x; i < n; i += gridDim.x * blockDim.x)
        p[i] = 0.f;
}

// ---------------------------------------------------------------------------
// Pre-swizzle weights into WMMA B-operand fragments.
// Fragment layout (per ISA 7.12.2, 16-bit 32x16 B operand, mirrored from the
// A 16x32 layout): lane l holds column N=(l&15); lanes>=16 take K offset +8;
// element e -> K = ktile*32 + (l>=16?8:0) + (e&7) + (e<8?0:16).
// B[k][n] = W[o = ntile*16+n][k]  (computes z = a @ W^T).
__global__ void k_prep(const float* W1, const float* W2, const float* W3, const float* W4,
                       _Float16* F1, _Float16* F2, _Float16* F3, _Float16* F4) {
    const float* Ws[4] = {W1, W2, W3, W4};
    _Float16*    Fs[4] = {F1, F2, F3, F4};
    const int fin[4]  = {DIN, HID, HID, HID};
    const int fout[4] = {HID, HID, HID, 3};
    const int kts[4]  = {8, 2, 2, 2};
    const int nts[4]  = {4, 4, 4, 1};
    for (int L = 0; L < 4; ++L) {
        int total = kts[L] * nts[L] * 512;
        for (int idx = threadIdx.x; idx < total; idx += blockDim.x) {
            int e    = idx & 15;
            int lane = (idx >> 4) & 31;
            int tile = idx >> 9;
            int nt   = tile % nts[L];
            int kt   = tile / nts[L];
            int k = kt * 32 + ((lane >= 16) ? 8 : 0) + (e & 7) + ((e < 8) ? 0 : 16);
            int o = nt * 16 + (lane & 15);
            Fs[L][idx] = (k < fin[L] && o < fout[L]) ? (_Float16)Ws[L][o * fin[L] + k]
                                                     : (_Float16)0.0f;
        }
    }
}

// ---------------------------------------------------------------------------
// Layer 1: z1 = lrelu(x @ W1^T + b1)  (K = 248 padded to 256), f16 output,
// fused group sum/sumsq/count accumulation.
// Block = 256 threads (8 waves); wave handles a 16-row x 64-col tile.
__global__ __launch_bounds__(256) void k_layer1(
    const float* __restrict__ x, const int* __restrict__ grp,
    const float* __restrict__ bias, const _Float16* __restrict__ Wf,
    _Float16* __restrict__ zout, float* __restrict__ stats, float* __restrict__ counts) {
    __shared__ __align__(16) _Float16 shW[8 * 4 * 512];   // 32 KB W1 fragments
    __shared__ __align__(16) _Float16 shZ[128 * HID];     // 16 KB z tile (row major)
    __shared__ float shS[2 * GROUPS * HID];               // 16 KB partial sum / sumsq
    __shared__ float shB[HID];
    __shared__ float shCnt[GROUPS];
    __shared__ int   shG[128];

    const int tid = threadIdx.x;
    for (int j = tid; j < 2048; j += 256) ((uint4*)shW)[j] = ((const uint4*)Wf)[j];
    if (tid < HID) shB[tid] = bias[tid];
    if (tid < 128) shG[tid] = grp[(long)blockIdx.x * 128 + tid];
    for (int j = tid; j < 2 * GROUPS * HID; j += 256) shS[j] = 0.f;
    if (tid < GROUPS) shCnt[tid] = 0.f;
    __syncthreads();

    const int wave = tid >> 5, lane = tid & 31;
    const int g16 = lane >> 4, l15 = lane & 15;
    const long row = (long)blockIdx.x * 128 + wave * 16 + l15;  // A-matrix row for this lane
    const float* xr = x + row * DIN;

    v8f acc[4] = {};
    for (int kt = 0; kt < 8; ++kt) {
        const int k0 = kt * 32 + g16 * 8;
        float4 a0 = *(const float4*)(xr + k0);
        float4 a1 = *(const float4*)(xr + k0 + 4);
        float4 a2 = make_float4(0.f, 0.f, 0.f, 0.f);
        float4 a3 = make_float4(0.f, 0.f, 0.f, 0.f);
        if (k0 + 16 < DIN) {  // only K=248..255 zero-pad tail fails this
            a2 = *(const float4*)(xr + k0 + 16);
            a3 = *(const float4*)(xr + k0 + 20);
        }
        v16h A;
        A[0] = (_Float16)a0.x;  A[1] = (_Float16)a0.y;  A[2] = (_Float16)a0.z;  A[3] = (_Float16)a0.w;
        A[4] = (_Float16)a1.x;  A[5] = (_Float16)a1.y;  A[6] = (_Float16)a1.z;  A[7] = (_Float16)a1.w;
        A[8] = (_Float16)a2.x;  A[9] = (_Float16)a2.y;  A[10] = (_Float16)a2.z; A[11] = (_Float16)a2.w;
        A[12] = (_Float16)a3.x; A[13] = (_Float16)a3.y; A[14] = (_Float16)a3.z; A[15] = (_Float16)a3.w;
#pragma unroll
        for (int nt = 0; nt < 4; ++nt) {
            v16h B = *(const v16h*)(shW + ((kt * 4 + nt) * 32 + lane) * 16);
            acc[nt] = __builtin_amdgcn_wmma_f32_16x16x32_f16(false, A, false, B,
                                                             (short)0, acc[nt], false, false);
        }
    }

    // Epilogue: bias + leaky-relu, D-layout -> row-major via LDS.
#pragma unroll
    for (int nt = 0; nt < 4; ++nt) {
        const int c = nt * 16 + l15;
        const float b = shB[c];
#pragma unroll
        for (int r = 0; r < 8; ++r) {
            const int zr = wave * 16 + r + g16 * 8;  // D layout: M = r + (lane<16 ? 0 : 8)
            shZ[zr * HID + c] = (_Float16)lrelu(acc[nt][r] + b);
        }
    }
    __syncthreads();

    // Fused segmented stats (sum, sumsq per [group][feature]) + counts.
    {
        const int d = tid & 63, q = tid >> 6;
        if (tid < 128) atomicAdd(&shCnt[shG[tid]], 1.0f);
        for (int r = q; r < 128; r += 4) {
            const float v = (float)shZ[r * HID + d];
            const int g = shG[r];
            atomicAdd(&shS[g * HID + d], v);
            atomicAdd(&shS[GROUPS * HID + g * HID + d], v * v);
        }
    }
    __syncthreads();
    for (int j = tid; j < 2 * GROUPS * HID; j += 256)
        if (shS[j] != 0.f) atomicAdd(&stats[j], shS[j]);
    if (tid < GROUPS && shCnt[tid] != 0.f) atomicAdd(&counts[tid], shCnt[tid]);

    // Coalesced b128 store of the f16 tile.
    const uint4* src = (const uint4*)shZ;
    uint4* dst = (uint4*)(zout + (long)blockIdx.x * 128 * HID);
    for (int j = tid; j < 1024; j += 256) dst[j] = src[j];
}

// ---------------------------------------------------------------------------
// Middle layers: z_out = lrelu(norm(z_in) @ W^T + b), K = 64, fused stats.
// norm params: [0][g][d] = mu, [1][g][d] = 1/(std+eps).
__global__ __launch_bounds__(256) void k_mid(
    const _Float16* __restrict__ zin, const int* __restrict__ grp,
    const float* __restrict__ bias, const _Float16* __restrict__ Wf,
    const float* __restrict__ norm,
    _Float16* __restrict__ zout, float* __restrict__ stats) {
    __shared__ __align__(16) _Float16 shW[2 * 4 * 512];   // 8 KB W fragments
    __shared__ __align__(16) _Float16 shZ[128 * HID];     // 16 KB z tile
    __shared__ float shN[2 * GROUPS * HID];               // 16 KB mu / istd
    __shared__ float shS[2 * GROUPS * HID];               // 16 KB partial stats
    __shared__ float shB[HID];
    __shared__ int   shG[128];

    const int tid = threadIdx.x;
    for (int j = tid; j < 512; j += 256) ((uint4*)shW)[j] = ((const uint4*)Wf)[j];
    for (int j = tid; j < 1024; j += 256) ((uint4*)shN)[j] = ((const uint4*)norm)[j];
    if (tid < HID) shB[tid] = bias[tid];
    if (tid < 128) shG[tid] = grp[(long)blockIdx.x * 128 + tid];
    for (int j = tid; j < 2 * GROUPS * HID; j += 256) shS[j] = 0.f;
    __syncthreads();

    const int wave = tid >> 5, lane = tid & 31;
    const int g16 = lane >> 4, l15 = lane & 15;
    const int lrow = wave * 16 + l15;
    const long row = (long)blockIdx.x * 128 + lrow;
    const int g = shG[lrow];
    const float* mu = shN + g * HID;
    const float* is = shN + GROUPS * HID + g * HID;
    const _Float16* zr = zin + row * HID;

    v8f acc[4] = {};
#pragma unroll
    for (int kt = 0; kt < 2; ++kt) {
        const int k0 = kt * 32 + g16 * 8;
        uint4 raw0 = *(const uint4*)(zr + k0);
        uint4 raw1 = *(const uint4*)(zr + k0 + 16);
        const _Float16* h0 = (const _Float16*)&raw0;
        const _Float16* h1 = (const _Float16*)&raw1;
        v16h A;
#pragma unroll
        for (int e = 0; e < 8; ++e) {
            A[e]     = (_Float16)(((float)h0[e] - mu[k0 + e])      * is[k0 + e]);
            A[8 + e] = (_Float16)(((float)h1[e] - mu[k0 + 16 + e]) * is[k0 + 16 + e]);
        }
#pragma unroll
        for (int nt = 0; nt < 4; ++nt) {
            v16h B = *(const v16h*)(shW + ((kt * 4 + nt) * 32 + lane) * 16);
            acc[nt] = __builtin_amdgcn_wmma_f32_16x16x32_f16(false, A, false, B,
                                                             (short)0, acc[nt], false, false);
        }
    }

#pragma unroll
    for (int nt = 0; nt < 4; ++nt) {
        const int c = nt * 16 + l15;
        const float b = shB[c];
#pragma unroll
        for (int r = 0; r < 8; ++r) {
            const int zrr = wave * 16 + r + g16 * 8;
            shZ[zrr * HID + c] = (_Float16)lrelu(acc[nt][r] + b);
        }
    }
    __syncthreads();

    {
        const int d = tid & 63, q = tid >> 6;
        for (int r = q; r < 128; r += 4) {
            const float v = (float)shZ[r * HID + d];
            const int gg = shG[r];
            atomicAdd(&shS[gg * HID + d], v);
            atomicAdd(&shS[GROUPS * HID + gg * HID + d], v * v);
        }
    }
    __syncthreads();
    for (int j = tid; j < 2 * GROUPS * HID; j += 256)
        if (shS[j] != 0.f) atomicAdd(&stats[j], shS[j]);

    const uint4* src = (const uint4*)shZ;
    uint4* dst = (uint4*)(zout + (long)blockIdx.x * 128 * HID);
    for (int j = tid; j < 1024; j += 256) dst[j] = src[j];
}

// ---------------------------------------------------------------------------
// (sum, sumsq, count) -> (mu, 1/(std+eps)) in place; ddof = 1 (torch.std).
__global__ void k_final(float* stats, const float* counts) {
    const int j = blockIdx.x * blockDim.x + threadIdx.x;
    if (j >= GROUPS * HID) return;
    const int g = j >> 6;
    const float cnt = counts[g];
    const float s = stats[j];
    const float sq = stats[GROUPS * HID + j];
    const float mean = s / fmaxf(cnt, 1.f);
    float var = (sq - cnt * mean * mean) / fmaxf(cnt - 1.f, 1.f);
    var = fmaxf(var, 0.f);
    stats[j] = mean;
    stats[GROUPS * HID + j] = 1.f / (sqrtf(var) + EPSI);
}

// ---------------------------------------------------------------------------
// Layer 4: out = norm(z3) @ W4^T + b4, 64 -> 3 (tiny: pure VALU, 1 row/thread).
__global__ __launch_bounds__(256) void k_out(
    const _Float16* __restrict__ zin, const int* __restrict__ grp,
    const float* __restrict__ norm, const float* __restrict__ W4,
    const float* __restrict__ b4, float* __restrict__ out, int Nrows) {
    __shared__ float shN[2 * GROUPS * HID];
    __shared__ float shW[3 * HID];
    __shared__ float shB[3];
    const int tid = threadIdx.x;
    for (int j = tid; j < 1024; j += 256) ((uint4*)shN)[j] = ((const uint4*)norm)[j];
    if (tid < 3 * HID) shW[tid] = W4[tid];
    if (tid < 3) shB[tid] = b4[tid];
    __syncthreads();

    const long n = (long)blockIdx.x * 256 + tid;
    if (n >= Nrows) return;
    const int g = grp[n];
    const float* mu = shN + g * HID;
    const float* is = shN + GROUPS * HID + g * HID;

    uint4 raw[8];
    const uint4* zp = (const uint4*)(zin + n * HID);
#pragma unroll
    for (int j = 0; j < 8; ++j) raw[j] = zp[j];
    const _Float16* h = (const _Float16*)raw;

    float a[HID];
#pragma unroll
    for (int d = 0; d < HID; ++d) a[d] = ((float)h[d] - mu[d]) * is[d];
#pragma unroll
    for (int o = 0; o < 3; ++o) {
        float s = shB[o];
#pragma unroll
        for (int d = 0; d < HID; ++d) s = fmaf(a[d], shW[o * HID + d], s);
        out[n * 3 + o] = s;
    }
}

// ---------------------------------------------------------------------------
extern "C" void kernel_launch(void* const* d_in, const int* in_sizes, int n_in,
                              void* d_out, int out_size, void* d_ws, size_t ws_size,
                              hipStream_t stream) {
    const float* x  = (const float*)d_in[0];
    const int*   gi = (const int*)d_in[1];
    const float* W1 = (const float*)d_in[2]; const float* b1 = (const float*)d_in[3];
    const float* W2 = (const float*)d_in[4]; const float* b2 = (const float*)d_in[5];
    const float* W3 = (const float*)d_in[6]; const float* b3 = (const float*)d_in[7];
    const float* W4 = (const float*)d_in[8]; const float* b4 = (const float*)d_in[9];

    const long Nr = (long)in_sizes[0] / DIN;  // 524288; multiple of 128

    char* ws = (char*)d_ws;
    size_t off = 0;
    auto alloc = [&](size_t bytes) -> void* {
        void* p = ws + off;
        off += (bytes + 255) & ~(size_t)255;
        return p;
    };
    _Float16* zA = (_Float16*)alloc((size_t)Nr * HID * 2);   // 64 MB
    _Float16* zB = (_Float16*)alloc((size_t)Nr * HID * 2);   // 64 MB
    _Float16* F1 = (_Float16*)alloc(8 * 4 * 512 * 2);
    _Float16* F2 = (_Float16*)alloc(2 * 4 * 512 * 2);
    _Float16* F3 = (_Float16*)alloc(2 * 4 * 512 * 2);
    _Float16* F4 = (_Float16*)alloc(2 * 1 * 512 * 2);
    float* st1  = (float*)alloc(2 * GROUPS * HID * 4);
    float* st2  = (float*)alloc(2 * GROUPS * HID * 4);
    float* st3  = (float*)alloc(2 * GROUPS * HID * 4);
    float* cnts = (float*)alloc(GROUPS * 4);

    // Zero all stat accumulators (st1..cnts are contiguous incl. pad).
    const int nzero = (int)(((char*)cnts - (char*)st1) / 4) + GROUPS;
    k_zero<<<32, 256, 0, stream>>>(st1, nzero);
    k_prep<<<1, 256, 0, stream>>>(W1, W2, W3, W4, F1, F2, F3, F4);

    const int nb = (int)(Nr / 128);
    k_layer1<<<nb, 256, 0, stream>>>(x, gi, b1, F1, zA, st1, cnts);
    k_final<<<(GROUPS * HID + 255) / 256, 256, 0, stream>>>(st1, cnts);
    k_mid<<<nb, 256, 0, stream>>>(zA, gi, b2, F2, st1, zB, st2);
    k_final<<<(GROUPS * HID + 255) / 256, 256, 0, stream>>>(st2, cnts);
    k_mid<<<nb, 256, 0, stream>>>(zB, gi, b3, F3, st2, zA, st3);
    k_final<<<(GROUPS * HID + 255) / 256, 256, 0, stream>>>(st3, cnts);

    const int nb2 = (int)((Nr + 255) / 256);
    k_out<<<nb2, 256, 0, stream>>>(zA, gi, st3, W4, b4, (float*)d_out, (int)Nr);
}
